// ScaleGNN_69312182223672
// MI455X (gfx1250) — matrix operand order repaired and checked by reading, not compiled, not run on hardware
//
#include <hip/hip_runtime.h>

// ---------------------------------------------------------------------------
// CDNA5 (gfx1250) wave32 WMMA types
// ---------------------------------------------------------------------------
typedef __bf16 v16bf __attribute__((ext_vector_type(16)));
typedef float  v8f   __attribute__((ext_vector_type(8)));

#define N_NODES 6144
#define C_IN    512
#define F_DIM   256
#define H_DIM   256
#define C_OUT   64
#define TOPK    10
#define EPSF    1e-8f
#define NBINS   16384

__device__ __forceinline__ unsigned short f2bf(float f) {
    union { float f; unsigned u; } v; v.f = f;
    unsigned r = v.u + 0x7FFFu + ((v.u >> 16) & 1u);   // round-to-nearest-even
    return (unsigned short)(r >> 16);
}

// ---------------------------------------------------------------------------
// Row L2 norms of features: one 256-thread block per row
// ---------------------------------------------------------------------------
__global__ void rownorm_kernel(const float* __restrict__ feat, float* __restrict__ norms) {
    __shared__ float red[256];
    const int row = blockIdx.x;
    float v = feat[(size_t)row * F_DIM + threadIdx.x];
    red[threadIdx.x] = v * v;
    __syncthreads();
    for (int s = 128; s > 0; s >>= 1) {
        if ((int)threadIdx.x < s) red[threadIdx.x] += red[threadIdx.x + s];
        __syncthreads();
    }
    if (threadIdx.x == 0) norms[row] = sqrtf(red[0]);
}

// ---------------------------------------------------------------------------
// f32 -> bf16 conversion (grid-stride)
// ---------------------------------------------------------------------------
__global__ void cvt_bf16_kernel(const float* __restrict__ in, unsigned short* __restrict__ out,
                                long long n) {
    long long i = (long long)blockIdx.x * blockDim.x + threadIdx.x;
    long long stride = (long long)gridDim.x * blockDim.x;
    for (; i < n; i += stride) out[i] = f2bf(in[i]);
}

// ---------------------------------------------------------------------------
// Histogram of mask in [0,1): LDS-local histogram (64 KB of 320 KB LDS),
// merged to global with atomics. One pass over 151 MB.
// ---------------------------------------------------------------------------
__global__ void zero_hist_kernel(unsigned int* hist) {
    hist[blockIdx.x * blockDim.x + threadIdx.x] = 0u;
}

__global__ void hist_kernel(const float* __restrict__ mask, unsigned int* __restrict__ hist,
                            long long n) {
    __shared__ unsigned int lh[NBINS];
    for (int i = threadIdx.x; i < NBINS; i += blockDim.x) lh[i] = 0u;
    __syncthreads();
    long long stride = (long long)gridDim.x * blockDim.x;
    for (long long i = (long long)blockIdx.x * blockDim.x + threadIdx.x; i < n; i += stride) {
        float v = mask[i];
        int b = (int)(v * (float)NBINS);
        b = b < 0 ? 0 : (b > NBINS - 1 ? NBINS - 1 : b);
        atomicAdd(&lh[b], 1u);
    }
    __syncthreads();
    for (int i = threadIdx.x; i < NBINS; i += blockDim.x)
        if (lh[i]) atomicAdd(&hist[i], lh[i]);
}

// p=0.5 quantile: scan histogram, interpolate linearly within the bin.
__global__ void threshold_kernel(const unsigned int* __restrict__ hist, float* __restrict__ th,
                                 long long n) {
    if (threadIdx.x == 0 && blockIdx.x == 0) {
        double target = 0.5 * (double)(n - 1);
        double cum = 0.0;
        int b = 0; unsigned int cnt = 0;
        for (b = 0; b < NBINS; ++b) {
            cnt = hist[b];
            if (cum + (double)cnt > target) break;
            cum += (double)cnt;
        }
        double frac = cnt ? (target - cum) / (double)cnt : 0.0;
        th[0] = (float)(((double)b + frac) / (double)NBINS);
    }
}

// ---------------------------------------------------------------------------
// Fused cosine-similarity GEMM + per-row top-K.
// One wave32 per 16-row stripe. A fragments for the stripe are staged once in
// LDS in WMMA fragment order; B fragments restaged per 16-column tile with
// next-stripe global_prefetch to hide latency. sim is never materialized
// (saves ~300 MB of HBM traffic).
// ---------------------------------------------------------------------------
__global__ void __launch_bounds__(32)
sim_topk_kernel(const unsigned short* __restrict__ feat16, const float* __restrict__ norms,
                int* __restrict__ topk_out) {
    __shared__ __align__(32) unsigned short Afrag[8][32][16];  // [kchunk][lane][elem]
    __shared__ __align__(32) unsigned short Bfrag[8][32][16];
    __shared__ float tile[16][17];
    __shared__ float topv[16][TOPK];
    __shared__ int   topi[16][TOPK];
    __shared__ float nA[16], nB[16];

    const int lane = threadIdx.x;
    const int hi = lane >> 4;
    const int lo = lane & 15;
    const int rowBase = blockIdx.x * 16;

    // Stage A stripe (16 rows x 256 K) into fragment layout (ISA 7.12.2, 16-bit A 16x32)
    for (int t = lane; t < 16 * (F_DIM / 8); t += 32) {
        int r = t / (F_DIM / 8);
        int kbase = (t % (F_DIM / 8)) * 8;
        const uint4 u = *(const uint4*)&feat16[(size_t)(rowBase + r) * F_DIM + kbase];
        const unsigned short* e8 = (const unsigned short*)&u;
#pragma unroll
        for (int j = 0; j < 8; ++j) {
            int k = kbase + j;
            int kc = k >> 5, k32 = k & 31;
            int ahi = (k32 >> 3) & 1;
            int ae  = (k32 & 7) + 8 * ((k32 >> 4) & 1);
            Afrag[kc][r + 16 * ahi][ae] = e8[j];
        }
    }
    if (lane < 16) {
        nA[lane] = norms[rowBase + lane];
#pragma unroll
        for (int q = 0; q < TOPK; ++q) { topv[lane][q] = -3.0e38f; topi[lane][q] = 0; }
    }
    __syncthreads();

    for (int cb = 0; cb < N_NODES; cb += 16) {
        // Stage B tile (B = F^T: B[k][n] = feat[cb+n][k]) into fragment layout
        for (int t = lane; t < 16 * (F_DIM / 8); t += 32) {
            int r = t / (F_DIM / 8);
            int kbase = (t % (F_DIM / 8)) * 8;
            const uint4 u = *(const uint4*)&feat16[(size_t)(cb + r) * F_DIM + kbase];
            const unsigned short* e8 = (const unsigned short*)&u;
#pragma unroll
            for (int j = 0; j < 8; ++j) {
                int k = kbase + j;
                int kc = k >> 5, k32 = k & 31;
                Bfrag[kc][r + 16 * ((k32 >> 3) & 1)][(k32 & 7) + 8 * (k32 >> 4)] = e8[j];
            }
        }
        if (lane < 16) nB[lane] = norms[cb + lane];
        // prefetch next column stripe while this one computes (global_prefetch_b8)
        if (cb + 16 < N_NODES) {
            __builtin_prefetch(&feat16[(size_t)(cb + 16 + lo) * F_DIM + hi * 128], 0, 0);
            __builtin_prefetch(&feat16[(size_t)(cb + 16 + lo) * F_DIM + 64 + hi * 128], 0, 0);
        }
        __syncthreads();

        v8f c = {};
#pragma unroll
        for (int kc = 0; kc < 8; ++kc) {
            v16bf a = *(const v16bf*)&Afrag[kc][lane][0];
            v16bf b = *(const v16bf*)&Bfrag[kc][lane][0];
            c = __builtin_amdgcn_wmma_f32_16x16x32_bf16(false, a, false, b,
                                                        (short)0, c, false, false);
        }
#pragma unroll
        for (int r = 0; r < 8; ++r) tile[r + 8 * hi][lo] = c[r];
        __syncthreads();

        if (lane < 16) {
            const float na = nA[lane];
            for (int j = 0; j < 16; ++j) {
                float s = tile[lane][j] / (na * nB[j] + EPSF);
                if (s > topv[lane][TOPK - 1]) {
                    int p = TOPK - 1;
                    while (p > 0 && topv[lane][p - 1] < s) {
                        topv[lane][p] = topv[lane][p - 1];
                        topi[lane][p] = topi[lane][p - 1];
                        --p;
                    }
                    topv[lane][p] = s;
                    topi[lane][p] = cb + j;
                }
            }
        }
        __syncthreads();
    }

    if (lane < 16) {
#pragma unroll
        for (int q = 0; q < TOPK; ++q)
            topk_out[(size_t)(rowBase + lane) * TOPK + q] = topi[lane][q];
    }
}

// ---------------------------------------------------------------------------
// A[i,j] = (j in topk[i]  ||  mask[i,j] > threshold) ? adj[i,j] : 0
// Writes f32 A (required output) + bf16 shadow for the GEMMs (halves re-read
// traffic; 75 MB fits the 192 MB L2).
// ---------------------------------------------------------------------------
__global__ void build_A_kernel(const float* __restrict__ adj, const float* __restrict__ mask,
                               const float* __restrict__ thp, const int* __restrict__ topk,
                               float* __restrict__ A, unsigned short* __restrict__ A16) {
    __shared__ int tk[TOPK];
    const int row = blockIdx.x;
    if ((int)threadIdx.x < TOPK) tk[threadIdx.x] = topk[(size_t)row * TOPK + threadIdx.x];
    __syncthreads();
    const float th = thp[0];
    const long long base = (long long)row * N_NODES;
    for (int j = threadIdx.x; j < N_NODES; j += blockDim.x) {
        bool hit = false;
#pragma unroll
        for (int q = 0; q < TOPK; ++q) hit = hit || (tk[q] == j);
        float a = (hit || (mask[base + j] > th)) ? adj[base + j] : 0.0f;
        A[base + j] = a;
        A16[base + j] = f2bf(a);
    }
}

// ---------------------------------------------------------------------------
// bf16 WMMA GEMM: C[M,Nn] = A[M,K] @ B[K,Nn] (+bias, optional relu).
// 256-thread block (8 waves) -> 64x32 C-block, each wave a 16x16 tile.
// K-tile 64 (two v_wmma_f32_16x16x32_bf16 per stage). Operands staged in LDS
// directly in fragment order -> 32 B contiguous per lane (ds_load_b128 x2).
// Next K-tile is prefetched (global_prefetch_b8) while this one computes.
// Outputs: f32 (Cf) and/or bf16 (Cb) for feeding the next GEMM.
// ---------------------------------------------------------------------------
__global__ void __launch_bounds__(256)
wmma_gemm_bf16(const unsigned short* __restrict__ A, const unsigned short* __restrict__ B,
               const float* __restrict__ bias,
               float* __restrict__ Cf, unsigned short* __restrict__ Cb,
               int Nn, int K, int relu) {
    __shared__ __align__(32) unsigned short Afrag[2][4][32][16]; // [khalf][mTile][lane][elem]
    __shared__ __align__(32) unsigned short Bfrag[2][2][32][16]; // [khalf][nTile][lane][elem]

    const int tid  = threadIdx.x;
    const int wave = tid >> 5, lane = tid & 31;
    const int hi = lane >> 4, lo = lane & 15;
    const int mTile = wave & 3, nTile = wave >> 2;
    const long long mBase = (long long)blockIdx.y * 64;
    const int nBase = blockIdx.x * 32;

    const int rA  = tid >> 2;            // A stage: row 0..63
    const int kbA = (tid & 3) * 16;      // A stage: k offset 0/16/32/48
    const int rB  = tid >> 2;            // B stage: k row 0..63
    const int nbB = (tid & 3) * 8;       // B stage: n offset 0/8/16/24

    v8f c = {};
    for (int k0 = 0; k0 < K; k0 += 64) {
        { // stage A: 64 rows x 64 k  (16 bf16 per thread)
            const uint4 u0 = *(const uint4*)&A[(mBase + rA) * (long long)K + k0 + kbA];
            const uint4 u1 = *(const uint4*)&A[(mBase + rA) * (long long)K + k0 + kbA + 8];
            const unsigned short* e0 = (const unsigned short*)&u0;
            const unsigned short* e1 = (const unsigned short*)&u1;
#pragma unroll
            for (int j = 0; j < 16; ++j) {
                int k = kbA + j;
                int kh = k >> 5, k32 = k & 31;
                int ahi = (k32 >> 3) & 1;
                int ae  = (k32 & 7) + 8 * ((k32 >> 4) & 1);
                Afrag[kh][rA >> 4][(rA & 15) + 16 * ahi][ae] = (j < 8) ? e0[j] : e1[j - 8];
            }
        }
        { // stage B: 64 k x 32 n  (8 bf16 per thread)
            const uint4 u = *(const uint4*)&B[(long long)(k0 + rB) * Nn + nBase + nbB];
            const unsigned short* e8 = (const unsigned short*)&u;
#pragma unroll
            for (int j = 0; j < 8; ++j) {
                int n  = nbB + j;
                int kh = rB >> 5, k32 = rB & 31;
                Bfrag[kh][n >> 4][(n & 15) + 16 * ((k32 >> 3) & 1)][(k32 & 7) + 8 * (k32 >> 4)] = e8[j];
            }
        }
        // prefetch next K-tile into cache while WMMAs run on this one
        if (k0 + 64 < K) {
            __builtin_prefetch(&A[(mBase + rA) * (long long)K + k0 + 64 + kbA], 0, 0);
            __builtin_prefetch(&B[(long long)(k0 + 64 + rB) * Nn + nBase + nbB], 0, 0);
        }
        __syncthreads();
#pragma unroll
        for (int kh = 0; kh < 2; ++kh) {
            v16bf a = *(const v16bf*)&Afrag[kh][mTile][lane][0];
            v16bf b = *(const v16bf*)&Bfrag[kh][nTile][lane][0];
            c = __builtin_amdgcn_wmma_f32_16x16x32_bf16(false, a, false, b,
                                                        (short)0, c, false, false);
        }
        __syncthreads();
    }

    const int col = nBase + nTile * 16 + lo;
    const float bb = bias ? bias[col] : 0.0f;
#pragma unroll
    for (int r = 0; r < 8; ++r) {
        long long row = mBase + mTile * 16 + r + 8 * hi;
        float v = c[r] + bb;
        if (relu) v = fmaxf(v, 0.0f);
        if (Cf) Cf[row * Nn + col] = v;
        if (Cb) Cb[row * Nn + col] = f2bf(v);
    }
}

// ---------------------------------------------------------------------------
// log_softmax over C_OUT=64 columns, one block per row
// ---------------------------------------------------------------------------
__global__ void logsoftmax_kernel(const float* __restrict__ h, float* __restrict__ out) {
    __shared__ float red[C_OUT];
    const int row = blockIdx.x;
    const int t = threadIdx.x;
    float v = h[(size_t)row * C_OUT + t];
    red[t] = v;
    __syncthreads();
    for (int s = 32; s > 0; s >>= 1) {
        if (t < s) red[t] = fmaxf(red[t], red[t + s]);
        __syncthreads();
    }
    const float mx = red[0];
    __syncthreads();
    red[t] = __expf(v - mx);
    __syncthreads();
    for (int s = 32; s > 0; s >>= 1) {
        if (t < s) red[t] += red[t + s];
        __syncthreads();
    }
    const float lse = mx + __logf(red[0]);
    out[(size_t)row * C_OUT + t] = v - lse;
}

// ---------------------------------------------------------------------------
// Launch
// ---------------------------------------------------------------------------
static inline size_t alignup(size_t x) { return (x + 255) & ~(size_t)255; }

extern "C" void kernel_launch(void* const* d_in, const int* in_sizes, int n_in,
                              void* d_out, int out_size, void* d_ws, size_t ws_size,
                              hipStream_t stream) {
    (void)in_sizes; (void)n_in; (void)out_size; (void)ws_size;
    const float* x    = (const float*)d_in[0]; // [N, C_IN]
    const float* adj  = (const float*)d_in[1]; // [N, N]
    const float* feat = (const float*)d_in[2]; // [N, F_DIM]
    const float* mask = (const float*)d_in[3]; // [N, N]
    const float* W1   = (const float*)d_in[4]; // [C_IN, H]
    const float* b1   = (const float*)d_in[5]; // [H]
    const float* W2   = (const float*)d_in[6]; // [H, C_OUT]
    const float* b2   = (const float*)d_in[7]; // [C_OUT]

    // Output tuple layout: [logsm N*C_OUT][A N*N][topk N*TOPK]
    float* outp  = (float*)d_out;
    float* logsm = outp;
    float* A     = outp + (size_t)N_NODES * C_OUT;
    int*   topk  = (int*)(outp + (size_t)N_NODES * C_OUT + (size_t)N_NODES * N_NODES);

    // Workspace carve-up
    char* ws = (char*)d_ws;
    float* norms        = (float*)ws;         ws += alignup(sizeof(float) * N_NODES);
    unsigned int* hist  = (unsigned int*)ws;  ws += alignup(sizeof(unsigned) * NBINS);
    float* th           = (float*)ws;         ws += 256;
    unsigned short* feat16 = (unsigned short*)ws; ws += alignup(2ull * N_NODES * F_DIM);
    unsigned short* x16    = (unsigned short*)ws; ws += alignup(2ull * N_NODES * C_IN);
    unsigned short* W116   = (unsigned short*)ws; ws += alignup(2ull * C_IN * H_DIM);
    unsigned short* W216   = (unsigned short*)ws; ws += alignup(2ull * H_DIM * C_OUT);
    unsigned short* A16    = (unsigned short*)ws; ws += alignup(2ull * N_NODES * N_NODES);
    unsigned short* Ax16   = (unsigned short*)ws; ws += alignup(2ull * N_NODES * C_IN);
    unsigned short* h116   = (unsigned short*)ws; ws += alignup(2ull * N_NODES * H_DIM);
    unsigned short* Ah16   = (unsigned short*)ws; ws += alignup(2ull * N_NODES * H_DIM);
    float* h2              = (float*)ws;          ws += alignup(sizeof(float) * N_NODES * C_OUT);

    const long long NN = (long long)N_NODES * N_NODES;

    // 1. row norms + bf16 conversions of static operands
    rownorm_kernel<<<N_NODES, 256, 0, stream>>>(feat, norms);
    cvt_bf16_kernel<<<1024, 256, 0, stream>>>(feat, feat16, (long long)N_NODES * F_DIM);
    cvt_bf16_kernel<<<1024, 256, 0, stream>>>(x, x16, (long long)N_NODES * C_IN);
    cvt_bf16_kernel<<<256, 256, 0, stream>>>(W1, W116, (long long)C_IN * H_DIM);
    cvt_bf16_kernel<<<64, 256, 0, stream>>>(W2, W216, (long long)H_DIM * C_OUT);

    // 2. p=0.5 quantile of mask via histogram
    zero_hist_kernel<<<NBINS / 256, 256, 0, stream>>>(hist);
    hist_kernel<<<1024, 256, 0, stream>>>(mask, hist, NN);
    threshold_kernel<<<1, 32, 0, stream>>>(hist, th, NN);

    // 3. fused cosine-sim WMMA GEMM + top-k (sim never materialized)
    sim_topk_kernel<<<N_NODES / 16, 32, 0, stream>>>(feat16, norms, topk);

    // 4. build filtered adjacency (f32 output + bf16 shadow, L2-resident)
    build_A_kernel<<<N_NODES, 256, 0, stream>>>(adj, mask, th, topk, A, A16);

    // 5. propagation: all GEMMs on v_wmma_f32_16x16x32_bf16
    dim3 blk(256);
    dim3 g1(C_IN / 32, N_NODES / 64);   // Ax = A @ x          [6144 x 512]
    wmma_gemm_bf16<<<g1, blk, 0, stream>>>(A16, x16, nullptr, nullptr, Ax16, C_IN, N_NODES, 0);
    dim3 g2(H_DIM / 32, N_NODES / 64);  // h1 = relu(Ax@W1+b1) [6144 x 256]
    wmma_gemm_bf16<<<g2, blk, 0, stream>>>(Ax16, W116, b1, nullptr, h116, H_DIM, C_IN, 1);
    dim3 g3(H_DIM / 32, N_NODES / 64);  // Ah = A @ h1         [6144 x 256]
    wmma_gemm_bf16<<<g3, blk, 0, stream>>>(A16, h116, nullptr, nullptr, Ah16, H_DIM, N_NODES, 0);
    dim3 g4(C_OUT / 32, N_NODES / 64);  // h2 = Ah@W2 + b2     [6144 x 64]
    wmma_gemm_bf16<<<g4, blk, 0, stream>>>(Ah16, W216, b2, h2, nullptr, C_OUT, H_DIM, 0);

    // 6. log-softmax
    logsoftmax_kernel<<<N_NODES, C_OUT, 0, stream>>>(h2, logsm);
}
